// DARN_23733989277862
// MI455X (gfx1250) — matrix operand order repaired and checked by reading, not compiled, not run on hardware
//
#include <hip/hip_runtime.h>
#include <hip/hip_bf16.h>

// ---------------- problem constants ----------------
#define B_     2048
#define C1_    256        // D/2
#define D_     512
#define S_     32
#define P_     784
#define KDEC   1295       // decoder K
#define KPAD   1312       // 41*32
#define NPAD   832        // 13*64, padded pixel dim for decoder B
#define M2_    100352     // B*7*7
#define K2_    4096       // 256*4*4
#define EPS_   1e-5f

typedef __attribute__((ext_vector_type(16))) _Float16 v16h;
typedef __attribute__((ext_vector_type(8)))  _Float16 v8h;
typedef __attribute__((ext_vector_type(8)))  float    v8f;

union AFrag { unsigned int u[8]; v16h h; };

__device__ __forceinline__ float log_sigmoid_f(float z) {
    // stable: min(z,0) - log1p(exp(-|z|))
    return fminf(z, 0.0f) - log1pf(__expf(-fabsf(z)));
}

__device__ __forceinline__ v16h join8(v8h lo, v8h hi) {
    v16h r;
#pragma unroll
    for (int i = 0; i < 8; ++i) { r[i] = lo[i]; r[8 + i] = hi[i]; }
    return r;
}

// 6x6 s2 p2 conv at (oh,ow) for channel weights in wloc
__device__ __forceinline__ float conv1_tap(const float* __restrict__ xp,
                                           const float* __restrict__ wloc,
                                           int oh, int ow, float bias) {
    float v = bias;
#pragma unroll
    for (int kh = 0; kh < 6; ++kh) {
        const int ih = oh * 2 - 2 + kh;
        if ((unsigned)ih < 28u) {
#pragma unroll
            for (int kw = 0; kw < 6; ++kw) {
                const int iw = ow * 2 - 2 + kw;
                if ((unsigned)iw < 28u) v = fmaf(xp[ih * 28 + iw], wloc[kh * 6 + kw], v);
            }
        }
    }
    return v;
}

// ============================================================
// conv1 pass 1: batch statistics only (training-mode BN1)
// ============================================================
__global__ __launch_bounds__(256) void k_conv1_stats(
    const float* __restrict__ x, const float* __restrict__ w1,
    const float* __restrict__ b1,
    float* __restrict__ bn1_sum, float* __restrict__ bn1_sq)
{
    __shared__ float wloc[36];
    __shared__ float rsum[256], rsq[256];
    const int c = blockIdx.x;
    const int chunk = blockIdx.y;     // 16 chunks of 128 batches
    const int tid = threadIdx.x;
    if (tid < 36) wloc[tid] = w1[c * 36 + tid];
    __syncthreads();

    const float bias = b1[c];
    float s = 0.f, q = 0.f;
    for (int i = tid; i < 128 * 196; i += 256) {
        const int bl = i / 196, hw = i % 196;
        const int b = chunk * 128 + bl;
        const float v = conv1_tap(x + (size_t)b * 784, wloc, hw / 14, hw % 14, bias);
        s += v; q += v * v;
    }
    rsum[tid] = s; rsq[tid] = q;
    __syncthreads();
    for (int off = 128; off; off >>= 1) {
        if (tid < off) { rsum[tid] += rsum[tid + off]; rsq[tid] += rsq[tid + off]; }
        __syncthreads();
    }
    if (tid == 0) { atomicAdd(&bn1_sum[c], rsum[0]); atomicAdd(&bn1_sq[c], rsq[0]); }
}

// ============================================================
// conv1 pass 2: recompute conv, apply BN1+ReLU, store f16 with
// zero-padded 16x16 spatial layout: h1p[c][b][16][16]
// (padded index = spatial index + 1; borders = conv2 zero-padding)
// ============================================================
__global__ __launch_bounds__(256) void k_conv1_store(
    const float* __restrict__ x, const float* __restrict__ w1,
    const float* __restrict__ b1,
    const float* __restrict__ scale1, const float* __restrict__ shift1,
    _Float16* __restrict__ h1p)
{
    __shared__ float wloc[36];
    const int c = blockIdx.x;
    const int chunk = blockIdx.y;
    const int tid = threadIdx.x;
    if (tid < 36) wloc[tid] = w1[c * 36 + tid];
    __syncthreads();

    const float bias = b1[c];
    const float sc = scale1[c], sh = shift1[c];
    for (int i = tid; i < 128 * 256; i += 256) {
        const int bl = i >> 8, pos = i & 255;
        const int b = chunk * 128 + bl;
        const int pr = pos >> 4, pc = pos & 15;       // padded coords
        float v = 0.f;
        if (pr >= 1 && pr <= 14 && pc >= 1 && pc <= 14) {
            const float cv = conv1_tap(x + (size_t)b * 784, wloc, pr - 1, pc - 1, bias);
            v = fmaxf(fmaf(cv, sc, sh), 0.f);
        }
        h1p[(((size_t)c * B_ + b) << 8) + pos] = (_Float16)v;
    }
}

// ============================================================
// fold BN(mean,var,gamma,beta) -> scale/shift
// ============================================================
__global__ void k_bnparam(const float* __restrict__ sum_, const float* __restrict__ sq_,
                          const float* __restrict__ g, const float* __restrict__ bt,
                          float* __restrict__ scale, float* __restrict__ shift,
                          int C, float invN)
{
    const int i = blockIdx.x * blockDim.x + threadIdx.x;
    if (i < C) {
        const float mean = sum_[i] * invN;
        const float var  = sq_[i] * invN - mean * mean;
        const float sc   = g[i] * rsqrtf(var + EPS_);
        scale[i] = sc;
        shift[i] = bt[i] - mean * sc;
    }
}

__global__ void k_f32tof16(const float* __restrict__ in, _Float16* __restrict__ out, int n)
{
    const int i = blockIdx.x * blockDim.x + threadIdx.x;
    if (i < n) out[i] = (_Float16)in[i];
}

// ============================================================
// conv2 as WMMA implicit GEMM: M=B*49, N=512, K=4096
//   A[m,k] gathered from pre-activated, zero-padded h1p -> branchless,
//          8 unconditional b32 loads per lane per K-step.
//   B[k,n] = conv2_w flattened OIHW = w2f16[n][k] row-major.
// Block: 256 thr = 8 waves, tile 128M x 128N, K-step 32.
// Epilogue: +bias, h2 f16 [N][M], fused BN2 stats.
// ============================================================
__global__ __launch_bounds__(256) void k_conv2_wmma(
    const _Float16* __restrict__ h1p,     // [256][B][256]  (16x16 padded)
    const _Float16* __restrict__ w2,      // [512][4096]
    const float* __restrict__ conv2_b,
    _Float16* __restrict__ h2,            // [512][M2]
    float* __restrict__ bn2_sum, float* __restrict__ bn2_sq)
{
    __shared__ float lsum[128], lsq[128];
    const int tid = threadIdx.x;
    if (tid < 128) { lsum[tid] = 0.f; lsq[tid] = 0.f; }
    __syncthreads();

    const int wv   = tid >> 5;
    const int lane = tid & 31;
    const int l15  = lane & 15;
    const int hi   = lane >> 4;          // 0/1
    const int m0   = blockIdx.x * 128;
    const int n0   = blockIdx.y * 128;

    // A row for this lane (fixed across the K loop)
    const int m  = m0 + wv * 16 + l15;
    const int b  = m / 49;
    const int hw = m % 49;
    const int oh = hw / 7, ow = hw % 7;
    // dword offset of the first 4-half run inside one padded image:
    //   halfs offset = (oh*2 + hi*2)*16 + ow*2  ->  dwords: 16*(oh+hi) + ow
    const int obase = 16 * (oh + hi) + ow;
    const unsigned int* __restrict__ h1u =
        (const unsigned int*)h1p + (size_t)b * 128 + obase;
    const size_t cstride = (size_t)B_ * 128;   // one channel, in dwords

    v8f acc[8];
#pragma unroll
    for (int i = 0; i < 8; ++i)
#pragma unroll
        for (int e = 0; e < 8; ++e) acc[i][e] = 0.f;

#pragma unroll 1
    for (int it = 0; it < 128; ++it) {   // K2/32
        const int kbase = it * 32;
        // ---- A fragment: channels c0=2*it (g=0) and c0+1 (g=1),
        //      rows kh = hi*2, hi*2+1; kw = 0..3 contiguous ----
        AFrag af;
        const unsigned int* p0 = h1u + (size_t)(2 * it) * cstride;
        const unsigned int* p1 = p0 + cstride;
        af.u[0] = p0[0]; af.u[1] = p0[1]; af.u[2] = p0[8]; af.u[3] = p0[9];
        af.u[4] = p1[0]; af.u[5] = p1[1]; af.u[6] = p1[8]; af.u[7] = p1[9];

#pragma unroll
        for (int nt = 0; nt < 8; ++nt) {
            const int n = n0 + nt * 16 + l15;
            const _Float16* bp = w2 + (size_t)n * K2_ + kbase + (hi ? 8 : 0);
            const v8h blo = *(const v8h*)bp;
            const v8h bhi = *(const v8h*)(bp + 16);
            acc[nt] = __builtin_amdgcn_wmma_f32_16x16x32_f16(
                false, af.h, false, join8(blo, bhi), (short)0, acc[nt], false, false);
        }
    }

    // epilogue: +bias, store f16 column-major, BN2 stats
#pragma unroll
    for (int nt = 0; nt < 8; ++nt) {
        const int n = n0 + nt * 16 + l15;
        const float bias = conv2_b[n];
        float s = 0.f, q = 0.f;
        _Float16* dst = h2 + (size_t)n * M2_ + m0 + wv * 16 + hi * 8;
#pragma unroll
        for (int r = 0; r < 8; ++r) {
            const float v = acc[nt][r] + bias;
            dst[r] = (_Float16)v;
            s += v; q += v * v;
        }
        atomicAdd(&lsum[nt * 16 + l15], s);
        atomicAdd(&lsq[nt * 16 + l15], q);
    }
    __syncthreads();
    if (tid < 128) {
        atomicAdd(&bn2_sum[n0 + tid], lsum[tid]);
        atomicAdd(&bn2_sq[n0 + tid],  lsq[tid]);
    }
}

// ============================================================
// BN2 + ReLU + 7x7 avg pool -> feat_enc [B][512] f32
// ============================================================
__global__ __launch_bounds__(256) void k_poolfeat(
    const _Float16* __restrict__ h2, const float* __restrict__ scale2,
    const float* __restrict__ shift2, float* __restrict__ feat_enc)
{
    const int b = blockIdx.x, tid = threadIdx.x;
    for (int d = tid; d < D_; d += 256) {
        const _Float16* p = h2 + (size_t)d * M2_ + b * 49;
        const float sc = scale2[d], sh = shift2[d];
        float s = 0.f;
#pragma unroll
        for (int i = 0; i < 49; ++i) s += fmaxf(fmaf((float)p[i], sc, sh), 0.f);
        feat_enc[b * D_ + d] = s * (1.f / 49.f);
    }
}

// ============================================================
// encoder logits + Bernoulli sample + log_posterior + AR prior + log_prior
// one wave (32 lanes) per batch row, lane = latent dim
// ============================================================
__global__ __launch_bounds__(32) void k_encode(
    const float* __restrict__ feat, const float* __restrict__ enc_w,
    const float* __restrict__ enc_b, const float* __restrict__ u,
    const float* __restrict__ prior_w, const float* __restrict__ prior_b,
    float* __restrict__ latent, float* __restrict__ logpost, float* __restrict__ logprior)
{
    const int b = blockIdx.x;
    const int s = threadIdx.x;   // 0..31
    const float* f = feat + (size_t)b * D_;
    const float* w = enc_w + (size_t)s * D_;
    float z = enc_b[s];
    for (int d = 0; d < D_; ++d) z = fmaf(f[d], w[d], z);

    const float sig = 1.f / (1.f + __expf(-z));
    const float lat = (u[b * S_ + s] < sig) ? 1.f : 0.f;
    const float sgn = 2.f * lat - 1.f;
    float lp = log_sigmoid_f(sgn * z);
    latent[b * S_ + s] = lat;

    // strictly-lower-triangular prior via wave shuffle
    float pz = prior_b[s];
#pragma unroll
    for (int j = 0; j < S_; ++j) {
        const float lj = __shfl(lat, j, 32);
        if (j < s) pz = fmaf(lj, prior_w[s * S_ + j], pz);
    }
    float lpr = log_sigmoid_f(sgn * pz);

#pragma unroll
    for (int off = 16; off; off >>= 1) {
        lp  += __shfl_xor(lp,  off, 32);
        lpr += __shfl_xor(lpr, off, 32);
    }
    if (s == 0) { logpost[b] = lp; logprior[b] = lpr; }
}

// ============================================================
// inv linear: feat_pre[b][d] = inv_b[d] + latent[b] . inv_w[d]
// ============================================================
__global__ __launch_bounds__(256) void k_inv(
    const float* __restrict__ latent, const float* __restrict__ inv_w,
    const float* __restrict__ inv_b, float* __restrict__ feat_pre)
{
    __shared__ float lat[S_];
    const int b = blockIdx.x, tid = threadIdx.x;
    if (tid < S_) lat[tid] = latent[b * S_ + tid];
    __syncthreads();
    for (int d = tid; d < D_; d += 256) {
        const float* w = inv_w + (size_t)d * S_;
        float a = inv_b[d];
#pragma unroll
        for (int s = 0; s < S_; ++s) a = fmaf(lat[s], w[s], a);
        feat_pre[(size_t)b * D_ + d] = a;
    }
}

// per-column stats over [rows][C] layout (for BN3)
__global__ __launch_bounds__(256) void k_colstats(
    const float* __restrict__ data, float* __restrict__ sum_, float* __restrict__ sq_,
    int rows, int C)
{
    __shared__ float rs[256], rq[256];
    const int d = blockIdx.x, tid = threadIdx.x;
    float s = 0.f, q = 0.f;
    for (int r = tid; r < rows; r += 256) {
        const float v = data[(size_t)r * C + d];
        s += v; q += v * v;
    }
    rs[tid] = s; rq[tid] = q;
    __syncthreads();
    for (int off = 128; off; off >>= 1) {
        if (tid < off) { rs[tid] += rs[tid + off]; rq[tid] += rq[tid + off]; }
        __syncthreads();
    }
    if (tid == 0) { sum_[d] = rs[0]; sq_[d] = rq[0]; }
}

// cat[b][0:512]=relu(bn3(feat_pre)); [512:1295]=x[b][0:783]; pad 0 to 1312; f16
__global__ __launch_bounds__(256) void k_buildcat(
    const float* __restrict__ feat_pre, const float* __restrict__ scale3,
    const float* __restrict__ shift3, const float* __restrict__ x,
    _Float16* __restrict__ cat)
{
    const int b = blockIdx.x, tid = threadIdx.x;
    for (int k = tid; k < KPAD; k += 256) {
        float v = 0.f;
        if (k < D_)        v = fmaxf(fmaf(feat_pre[(size_t)b * D_ + k], scale3[k], shift3[k]), 0.f);
        else if (k < KDEC) v = x[(size_t)b * P_ + (k - D_)];
        cat[(size_t)b * KPAD + k] = (_Float16)v;
    }
}

// decoder weights: pre-masked f16 [NPAD][KPAD]; mask: k <= 511 + n
__global__ void k_maskdec(const float* __restrict__ dec_w, _Float16* __restrict__ dw)
{
    const int i = blockIdx.x * blockDim.x + threadIdx.x;
    if (i >= NPAD * KPAD) return;
    const int n = i / KPAD, k = i % KPAD;
    float v = 0.f;
    if (n < P_ && k < KDEC && k <= (D_ - 1 + n)) v = dec_w[(size_t)n * KDEC + k];
    dw[i] = (_Float16)v;
}

// ============================================================
// decoder WMMA GEMM M=2048,N=784(pad 832),K=1312 + fused BCE epilogue
// Block 256 thr (8 waves), tile 128M x 64N.
// ============================================================
__global__ __launch_bounds__(256) void k_dec_wmma(
    const _Float16* __restrict__ cat,   // [2048][KPAD]
    const _Float16* __restrict__ dw,    // [NPAD][KPAD]
    const float* __restrict__ dec_b, const float* __restrict__ x,
    float* __restrict__ logsamp)
{
    __shared__ float lsamp[128];
    const int tid = threadIdx.x;
    if (tid < 128) lsamp[tid] = 0.f;
    __syncthreads();

    const int wv   = tid >> 5;
    const int lane = tid & 31;
    const int l15  = lane & 15;
    const int hi   = lane >> 4;
    const int m0   = blockIdx.x * 128;
    const int n0   = blockIdx.y * 64;

    const int m = m0 + wv * 16 + l15;
    const _Float16* arow = cat + (size_t)m * KPAD + (hi ? 8 : 0);

    v8f acc[4];
#pragma unroll
    for (int i = 0; i < 4; ++i)
#pragma unroll
        for (int e = 0; e < 8; ++e) acc[i][e] = 0.f;

#pragma unroll 1
    for (int it = 0; it < KPAD / 32; ++it) {
        const int kbase = it * 32;
        const v8h alo = *(const v8h*)(arow + kbase);
        const v8h ahi = *(const v8h*)(arow + kbase + 16);
        const v16h a = join8(alo, ahi);
#pragma unroll
        for (int nt = 0; nt < 4; ++nt) {
            const int n = n0 + nt * 16 + l15;
            const _Float16* bp = dw + (size_t)n * KPAD + kbase + (hi ? 8 : 0);
            const v8h blo = *(const v8h*)bp;
            const v8h bhi = *(const v8h*)(bp + 16);
            acc[nt] = __builtin_amdgcn_wmma_f32_16x16x32_f16(
                false, a, false, join8(blo, bhi), (short)0, acc[nt], false, false);
        }
    }

    // fused BCE-with-logits epilogue -> per-batch partial sums
#pragma unroll
    for (int nt = 0; nt < 4; ++nt) {
        const int n = n0 + nt * 16 + l15;    // pixel index
        if (n < P_) {
            const float bias = dec_b[n];
#pragma unroll
            for (int r = 0; r < 8; ++r) {
                const int ml = wv * 16 + hi * 8 + r;     // batch within block tile
                const float z  = acc[nt][r] + bias;
                const float xv = x[(size_t)(m0 + ml) * P_ + n];
                const float t  = xv * log_sigmoid_f(z) + (1.f - xv) * log_sigmoid_f(-z);
                atomicAdd(&lsamp[ml], t);
            }
        }
    }
    __syncthreads();
    if (tid < 128) atomicAdd(&logsamp[m0 + tid], lsamp[tid]);
}

// final: mean over batch of (log_posterior - log_prior - log_sampling)
__global__ __launch_bounds__(256) void k_final(
    const float* __restrict__ lp, const float* __restrict__ lpr,
    const float* __restrict__ ls, float* __restrict__ out)
{
    __shared__ float r[256];
    const int tid = threadIdx.x;
    float s = 0.f;
    for (int b = tid; b < B_; b += 256) s += lp[b] - lpr[b] - ls[b];
    r[tid] = s;
    __syncthreads();
    for (int off = 128; off; off >>= 1) {
        if (tid < off) r[tid] += r[tid + off];
        __syncthreads();
    }
    if (tid == 0) out[0] = r[0] * (1.f / (float)B_);
}

// ============================================================
extern "C" void kernel_launch(void* const* d_in, const int* in_sizes, int n_in,
                              void* d_out, int out_size, void* d_ws, size_t ws_size,
                              hipStream_t stream) {
    (void)in_sizes; (void)n_in; (void)out_size; (void)ws_size;
    const float* x       = (const float*)d_in[0];
    const float* u       = (const float*)d_in[1];
    const float* conv1_w = (const float*)d_in[2];
    const float* conv1_b = (const float*)d_in[3];
    const float* bn1_g   = (const float*)d_in[4];
    const float* bn1_b   = (const float*)d_in[5];
    const float* conv2_w = (const float*)d_in[6];
    const float* conv2_b = (const float*)d_in[7];
    const float* bn2_g   = (const float*)d_in[8];
    const float* bn2_b   = (const float*)d_in[9];
    const float* enc_w   = (const float*)d_in[10];
    const float* enc_b   = (const float*)d_in[11];
    const float* inv_w   = (const float*)d_in[12];
    const float* inv_b   = (const float*)d_in[13];
    const float* bn3_g   = (const float*)d_in[14];
    const float* bn3_b   = (const float*)d_in[15];
    const float* prior_w = (const float*)d_in[16];
    const float* prior_b = (const float*)d_in[17];
    const float* dec_w   = (const float*)d_in[18];
    const float* dec_b   = (const float*)d_in[19];
    float* out = (float*)d_out;

    char* ws = (char*)d_ws;
    size_t off = 0;
    auto alloc = [&](size_t bytes) -> void* {
        void* p = ws + off;
        off += (bytes + 255) & ~(size_t)255;
        return p;
    };
    _Float16* h1p      = (_Float16*)alloc((size_t)C1_ * B_ * 256 * sizeof(_Float16));
    _Float16* w2f      = (_Float16*)alloc((size_t)D_ * K2_ * sizeof(_Float16));
    _Float16* h2f      = (_Float16*)alloc((size_t)D_ * M2_ * sizeof(_Float16));
    _Float16* catf     = (_Float16*)alloc((size_t)B_ * KPAD * sizeof(_Float16));
    _Float16* dwf      = (_Float16*)alloc((size_t)NPAD * KPAD * sizeof(_Float16));
    float* feat_pre    = (float*)alloc((size_t)B_ * D_ * sizeof(float));
    float* feat_enc    = (float*)alloc((size_t)B_ * D_ * sizeof(float));
    float* latent      = (float*)alloc((size_t)B_ * S_ * sizeof(float));
    float* logpost     = (float*)alloc(B_ * sizeof(float));
    float* logprior    = (float*)alloc(B_ * sizeof(float));
    float* logsamp     = (float*)alloc(B_ * sizeof(float));
    float* bn1_sum     = (float*)alloc(C1_ * sizeof(float));
    float* bn1_sq      = (float*)alloc(C1_ * sizeof(float));
    float* bn2_sum     = (float*)alloc(D_ * sizeof(float));
    float* bn2_sq      = (float*)alloc(D_ * sizeof(float));
    float* bn3_sum     = (float*)alloc(D_ * sizeof(float));
    float* bn3_sq      = (float*)alloc(D_ * sizeof(float));
    float* scale1      = (float*)alloc(C1_ * sizeof(float));
    float* shift1      = (float*)alloc(C1_ * sizeof(float));
    float* scale2      = (float*)alloc(D_ * sizeof(float));
    float* shift2      = (float*)alloc(D_ * sizeof(float));
    float* scale3      = (float*)alloc(D_ * sizeof(float));
    float* shift3      = (float*)alloc(D_ * sizeof(float));

    // zero atomic accumulators (deterministic every call)
    hipMemsetAsync(logsamp, 0, B_ * sizeof(float), stream);
    hipMemsetAsync(bn1_sum, 0, C1_ * sizeof(float), stream);
    hipMemsetAsync(bn1_sq,  0, C1_ * sizeof(float), stream);
    hipMemsetAsync(bn2_sum, 0, D_ * sizeof(float), stream);
    hipMemsetAsync(bn2_sq,  0, D_ * sizeof(float), stream);

    // 1) conv1 stats -> BN1 fold -> conv1 store (BN+ReLU pre-applied, padded)
    k_conv1_stats<<<dim3(C1_, 16), 256, 0, stream>>>(x, conv1_w, conv1_b, bn1_sum, bn1_sq);
    k_bnparam<<<1, 256, 0, stream>>>(bn1_sum, bn1_sq, bn1_g, bn1_b, scale1, shift1,
                                     C1_, 1.f / (float)(B_ * 196));
    k_conv1_store<<<dim3(C1_, 16), 256, 0, stream>>>(x, conv1_w, conv1_b, scale1, shift1, h1p);
    // 2) pack conv2 weights to f16 [N][K]
    k_f32tof16<<<(D_ * K2_ + 255) / 256, 256, 0, stream>>>(conv2_w, w2f, D_ * K2_);
    // 3) conv2 implicit GEMM (WMMA) + BN2 stats
    k_conv2_wmma<<<dim3(M2_ / 128, D_ / 128), 256, 0, stream>>>(
        h1p, w2f, conv2_b, h2f, bn2_sum, bn2_sq);
    k_bnparam<<<2, 256, 0, stream>>>(bn2_sum, bn2_sq, bn2_g, bn2_b, scale2, shift2,
                                     D_, 1.f / (float)M2_);
    // 4) BN2+ReLU+avgpool
    k_poolfeat<<<B_, 256, 0, stream>>>(h2f, scale2, shift2, feat_enc);
    // 5) encode + sample + posterior + AR prior
    k_encode<<<B_, 32, 0, stream>>>(feat_enc, enc_w, enc_b, u, prior_w, prior_b,
                                    latent, logpost, logprior);
    // 6) inv linear + BN3
    k_inv<<<B_, 256, 0, stream>>>(latent, inv_w, inv_b, feat_pre);
    k_colstats<<<D_, 256, 0, stream>>>(feat_pre, bn3_sum, bn3_sq, B_, D_);
    k_bnparam<<<2, 256, 0, stream>>>(bn3_sum, bn3_sq, bn3_g, bn3_b, scale3, shift3,
                                     D_, 1.f / (float)B_);
    // 7) build cat (f16) and masked decoder weights (f16)
    k_buildcat<<<B_, 256, 0, stream>>>(feat_pre, scale3, shift3, x, catf);
    k_maskdec<<<(NPAD * KPAD + 255) / 256, 256, 0, stream>>>(dec_w, dwf);
    // 8) decoder GEMM (WMMA) + fused BCE
    k_dec_wmma<<<dim3(B_ / 128, NPAD / 64), 256, 0, stream>>>(catf, dwf, dec_b, x, logsamp);
    // 9) final scalar
    k_final<<<1, 256, 0, stream>>>(logpost, logprior, logsamp, out);
}
